// MultiHeadAttention_63531156242710
// MI455X (gfx1250) — matrix operand order repaired and checked by reading, not compiled, not run on hardware
//
#include <hip/hip_runtime.h>
#include <hip/hip_bf16.h>
#include <math.h>

// ---------------- types ----------------
typedef __bf16 bf16_t;
typedef __attribute__((ext_vector_type(16))) bf16_t v16bf;
typedef __attribute__((ext_vector_type(8)))  float  v8f;
typedef __attribute__((__vector_size__(16))) int v4i_t;

struct __attribute__((aligned(16))) U128 { unsigned int a, b, c, d; };
struct __attribute__((aligned(16))) F128 { float x, y, z, w; };
struct __attribute__((aligned(8)))  U64  { unsigned int x, y; };

union FragB { v16bf v; U128 u[2]; unsigned short h[16]; };
union AccF  { v8f v; float f[8]; };
union V128U { U128 u; unsigned short s[8]; };

#define N_HEADS  6
#define D_KV     64
#define D_MODEL  1472
#define INNER    (N_HEADS * D_KV)   // 384
#define BATCH    4
#define SEQ      2048

#define BM 64
#define BN 64
#define BK 32
#define LDA 40      // LDS row stride (elements) for 32-wide tiles; mult of 8
#define LDK 72      // LDS row stride (elements) for 64-wide tiles; mult of 8

// ---------------- async global->LDS (gfx1250 ASYNCcnt path) ----------------
#if defined(__AMDGCN__) && __has_builtin(__builtin_amdgcn_global_load_async_to_lds_b128) && __has_builtin(__builtin_amdgcn_s_wait_asynccnt)
#define HAVE_ASYNC_LDS 1
#else
#define HAVE_ASYNC_LDS 0
#endif

__device__ inline void async_cp_b128(const void* g, void* l) {
#if HAVE_ASYNC_LDS
  // Builtin expects: (v4i addrspace(1)* gsrc, v4i addrspace(3)* ldst, imm offset, imm cpol).
  // Rebuild address-space pointers through integers: low 32 bits of a generic
  // LDS address are the LDS byte offset; global generic == AS1 value.
  typedef __attribute__((address_space(1))) v4i_t* gptr_t;
  typedef __attribute__((address_space(3))) v4i_t* lptr_t;
  gptr_t gp = (gptr_t)(unsigned long long)g;
  lptr_t lp = (lptr_t)(unsigned int)(unsigned long long)l;
  __builtin_amdgcn_global_load_async_to_lds_b128(gp, lp, 0, 0);
#else
  *(U128*)l = *(const U128*)g;
#endif
}

__device__ inline void async_wait_all() {
#if HAVE_ASYNC_LDS
  __builtin_amdgcn_s_wait_asynccnt(0);
#endif
}

// ---------------- helpers ----------------
__device__ inline unsigned short f32_to_bf16_rn(float x) {
  union { float f; unsigned int u; } c; c.f = x;
  unsigned int u = c.u;
  u += 0x7FFFu + ((u >> 16) & 1u);
  return (unsigned short)(u >> 16);
}

__device__ inline v8f wmma_bf16(v16bf a, v16bf b, v8f c) {
  return __builtin_amdgcn_wmma_f32_16x16x32_bf16(false, a, false, b, (short)0, c,
                                                 false, false);
}

// A-matrix fragment (16x32 bf16) from row-major LDS tile.
// Lanes 0-15: row=lane,    K = k0+0..7   (u[0]) and k0+16..23 (u[1])
// Lanes 16-31: row=lane-16, K = k0+8..15 (u[0]) and k0+24..31 (u[1])
__device__ inline v16bf load_A_frag(const unsigned short* lds, int row0, int k0,
                                    int stride, int lane) {
  int row = row0 + (lane & 15);
  int kb  = k0 + ((lane >> 4) << 3);
  FragB f;
  const unsigned short* p = lds + row * stride + kb;
  f.u[0] = *(const U128*)(p);
  f.u[1] = *(const U128*)(p + 16);
  return f.v;
}

// B-matrix fragment (32x16 bf16) from LDS holding B^T row-major ([n][k]).
// Lanes 0-15: n=lane,    K = k0+0..7, k0+16..23 ; lanes 16-31: n=lane-16, K = k0+8..15, k0+24..31
__device__ inline v16bf load_Bt_frag(const unsigned short* lds, int n0, int k0,
                                     int stride, int lane) {
  int n  = n0 + (lane & 15);
  int kb = k0 + ((lane >> 4) << 3);
  FragB f;
  const unsigned short* p = lds + n * stride + kb;
  f.u[0] = *(const U128*)(p);
  f.u[1] = *(const U128*)(p + 16);
  return f.v;
}

// T5 bidirectional relative-position bucket (32 buckets, max_distance 128)
__device__ inline int rel_bucket(int rel) {
  int ret = (rel > 0) ? 16 : 0;
  int n = (rel < 0) ? -rel : rel;
  if (n < 8) return ret + n;
  // 8 + log(n/8)/log(128/8) * 8
  int large = 8 + (int)(__logf((float)n * 0.125f) * (8.0f / __logf(16.0f)));
  if (large > 15) large = 15;
  return ret + large;
}

// ============================================================
// Kernel 1: fused QKV projection.  Q/K/V = x @ Wq/Wk/Wv
// x: [8192, 1472] f32, W: [1472, 384] f32 -> bf16 [B*H, S, 64]
// ============================================================
__global__ void __launch_bounds__(128)
qkv_kernel(const float* __restrict__ x, const float* __restrict__ Wq,
           const float* __restrict__ Wk, const float* __restrict__ Wv,
           unsigned short* __restrict__ Qb, unsigned short* __restrict__ Kb,
           unsigned short* __restrict__ Vb) {
  __shared__ unsigned short As[BM * LDA];
  __shared__ unsigned short Bt[3][BN * LDA];

  const int tid  = threadIdx.x;
  const int lane = tid & 31;
  const int wave = tid >> 5;
  const int wm = (wave >> 1) * 32;   // wave row quadrant
  const int wn = (wave & 1) * 32;    // wave col quadrant
  const int m0 = blockIdx.y * BM;
  const int n0 = blockIdx.x * BN;

  const float* const Wp[3] = {Wq, Wk, Wv};
  unsigned short* const Dp[3] = {Qb, Kb, Vb};

  AccF acc[3][2][2];
#pragma unroll
  for (int wi = 0; wi < 3; ++wi)
#pragma unroll
    for (int ti = 0; ti < 2; ++ti)
#pragma unroll
      for (int tj = 0; tj < 2; ++tj)
#pragma unroll
        for (int i = 0; i < 8; ++i) acc[wi][ti][tj].f[i] = 0.0f;

  for (int k0 = 0; k0 < D_MODEL; k0 += BK) {
    __syncthreads();
    // prefetch next K-tile (fp32 paths must convert through VGPRs)
    if (k0 + BK < D_MODEL) {
      __builtin_prefetch(&x[(size_t)(m0 + (tid >> 1)) * D_MODEL + k0 + BK], 0, 3);
      __builtin_prefetch(&Wp[tid % 3][(size_t)(k0 + BK + (tid >> 2)) * INNER + n0], 0, 3);
    }
    // stage A tile (64x32 f32 -> bf16), 512 float4 loads across 128 threads
#pragma unroll
    for (int it = 0; it < 4; ++it) {
      int i = tid + it * 128;
      int r  = i >> 3;             // 8 float4 per row
      int c4 = (i & 7) << 2;
      F128 f = *(const F128*)&x[(size_t)(m0 + r) * D_MODEL + k0 + c4];
      U64 pk;
      pk.x = (unsigned)f32_to_bf16_rn(f.x) | ((unsigned)f32_to_bf16_rn(f.y) << 16);
      pk.y = (unsigned)f32_to_bf16_rn(f.z) | ((unsigned)f32_to_bf16_rn(f.w) << 16);
      *(U64*)&As[r * LDA + c4] = pk;
    }
    // stage B^T tiles for all 3 weights (32x64 f32 -> 64x32 bf16 transposed)
#pragma unroll
    for (int wi = 0; wi < 3; ++wi) {
#pragma unroll
      for (int it = 0; it < 4; ++it) {
        int i = tid + it * 128;
        int k  = i >> 4;            // 16 float4 per k-row
        int n4 = (i & 15) << 2;
        F128 f = *(const F128*)&Wp[wi][(size_t)(k0 + k) * INNER + n0 + n4];
        Bt[wi][(n4 + 0) * LDA + k] = f32_to_bf16_rn(f.x);
        Bt[wi][(n4 + 1) * LDA + k] = f32_to_bf16_rn(f.y);
        Bt[wi][(n4 + 2) * LDA + k] = f32_to_bf16_rn(f.z);
        Bt[wi][(n4 + 3) * LDA + k] = f32_to_bf16_rn(f.w);
      }
    }
    __syncthreads();

    v16bf af[2];
    af[0] = load_A_frag(As, wm + 0,  0, LDA, lane);
    af[1] = load_A_frag(As, wm + 16, 0, LDA, lane);
#pragma unroll
    for (int wi = 0; wi < 3; ++wi)
#pragma unroll
      for (int tj = 0; tj < 2; ++tj) {
        v16bf bf = load_Bt_frag(Bt[wi], wn + tj * 16, 0, LDA, lane);
#pragma unroll
        for (int ti = 0; ti < 2; ++ti)
          acc[wi][ti][tj].v = wmma_bf16(af[ti], bf, acc[wi][ti][tj].v);
      }
  }

  // epilogue: C layout -> [B,H,S,64] bf16
  const int mloc0 = (lane >> 4) << 3;
  const int nloc  = lane & 15;
#pragma unroll
  for (int wi = 0; wi < 3; ++wi)
#pragma unroll
    for (int ti = 0; ti < 2; ++ti)
#pragma unroll
      for (int tj = 0; tj < 2; ++tj)
#pragma unroll
        for (int i = 0; i < 8; ++i) {
          int m = m0 + wm + ti * 16 + mloc0 + i;
          int n = n0 + wn + tj * 16 + nloc;
          int b = m >> 11, s = m & 2047;
          int h = n >> 6,  d = n & 63;
          size_t off = ((size_t)(b * N_HEADS + h) * SEQ + s) * D_KV + d;
          Dp[wi][off] = f32_to_bf16_rn(acc[wi][ti][tj].f[i]);
        }
}

// ============================================================
// Kernel 2: flash attention with T5 relative-position bias.
// Per block: 64 query rows of one (b,h). bf16 WMMA, fp32 softmax state.
// Q/K tiles staged with GLOBAL_LOAD_ASYNC_TO_LDS_B128 (ASYNCcnt).
// ============================================================
__global__ void __launch_bounds__(128)
attn_kernel(const unsigned short* __restrict__ Qb,
            const unsigned short* __restrict__ Kb,
            const unsigned short* __restrict__ Vb,
            const float* __restrict__ rel_emb,
            unsigned short* __restrict__ Ob) {
  __shared__ unsigned short Qs[64 * LDK];
  __shared__ unsigned short Ks[64 * LDK];
  __shared__ unsigned short Vt[64 * LDK];       // V transposed: [dim][key]
  __shared__ unsigned short Ps[4][16 * LDK];    // per-wave P tile
  __shared__ float biasTab[4096];

  const int tid  = threadIdx.x;
  const int lane = tid & 31;
  const int wave = tid >> 5;
  const int bh = blockIdx.x;            // b*N_HEADS + h
  const int h  = bh % N_HEADS;
  const int b  = bh / N_HEADS;
  const int q0 = blockIdx.y * 64;

  const unsigned short* Qp = Qb + (size_t)bh * SEQ * D_KV;
  const unsigned short* Kp = Kb + (size_t)bh * SEQ * D_KV;
  const unsigned short* Vp = Vb + (size_t)bh * SEQ * D_KV;

  // resident Q tile (64x64 bf16) via async DMA to LDS
#pragma unroll
  for (int it = 0; it < 4; ++it) {
    int i = tid + it * 128;
    int r = i >> 3, c8 = (i & 7) << 3;
    async_cp_b128(&Qp[(size_t)(q0 + r) * D_KV + c8], &Qs[r * LDK + c8]);
  }

  // bias table: bias(rel = k - q), rel in [-2048, 2047]
  for (int i = tid; i < 4096; i += 128) {
    int rel = i - 2048;
    biasTab[i] = rel_emb[rel_bucket(rel) * N_HEADS + h];
  }

  AccF o[4];
  float rm[8], rl[8];
#pragma unroll
  for (int j = 0; j < 4; ++j)
#pragma unroll
    for (int i = 0; i < 8; ++i) o[j].f[i] = 0.0f;
#pragma unroll
  for (int i = 0; i < 8; ++i) { rm[i] = -1e30f; rl[i] = 0.0f; }

  const int mloc0 = (lane >> 4) << 3;
  const int nloc  = lane & 15;

  for (int kb = 0; kb < SEQ / 64; ++kb) {
    __syncthreads();
    // stage K via async DMA (row-major == B^T layout for Q*K^T); V^T via VGPRs
#pragma unroll
    for (int it = 0; it < 4; ++it) {
      int i = tid + it * 128;
      int r = i >> 3, c8 = (i & 7) << 3;
      async_cp_b128(&Kp[(size_t)(kb * 64 + r) * D_KV + c8], &Ks[r * LDK + c8]);
      V128U vv;
      vv.u = *(const U128*)&Vp[(size_t)(kb * 64 + r) * D_KV + c8];
#pragma unroll
      for (int j = 0; j < 8; ++j) Vt[(c8 + j) * LDK + r] = vv.s[j];
    }
    async_wait_all();   // Q (first iter) + K landed in LDS
    __syncthreads();

    // scores S = Q K^T  (16x64 strip per wave)
    AccF sc[4];
#pragma unroll
    for (int j = 0; j < 4; ++j)
#pragma unroll
      for (int i = 0; i < 8; ++i) sc[j].f[i] = 0.0f;
#pragma unroll
    for (int kt = 0; kt < 2; ++kt) {
      v16bf a = load_A_frag(Qs, wave * 16, kt * 32, LDK, lane);
#pragma unroll
      for (int j = 0; j < 4; ++j) {
        v16bf bb = load_Bt_frag(Ks, j * 16, kt * 32, LDK, lane);
        sc[j].v = wmma_bf16(a, bb, sc[j].v);
      }
    }

    // bias + online softmax (rows live across 16-lane halves)
#pragma unroll
    for (int i = 0; i < 8; ++i) {
      int qg = q0 + wave * 16 + mloc0 + i;
      float mx = -1e30f;
#pragma unroll
      for (int j = 0; j < 4; ++j) {
        int kg = kb * 64 + j * 16 + nloc;
        sc[j].f[i] += biasTab[kg - qg + 2048];
        mx = fmaxf(mx, sc[j].f[i]);
      }
      mx = fmaxf(mx, __shfl_xor(mx, 1));
      mx = fmaxf(mx, __shfl_xor(mx, 2));
      mx = fmaxf(mx, __shfl_xor(mx, 4));
      mx = fmaxf(mx, __shfl_xor(mx, 8));

      float mnew  = fmaxf(rm[i], mx);
      float scale = __expf(rm[i] - mnew);
      rm[i] = mnew;
      float ps = 0.0f;
#pragma unroll
      for (int j = 0; j < 4; ++j) {
        float p = __expf(sc[j].f[i] - mnew);
        ps += p;
        Ps[wave][(mloc0 + i) * LDK + j * 16 + nloc] = f32_to_bf16_rn(p);
      }
      ps += __shfl_xor(ps, 1);
      ps += __shfl_xor(ps, 2);
      ps += __shfl_xor(ps, 4);
      ps += __shfl_xor(ps, 8);
      rl[i] = rl[i] * scale + ps;
#pragma unroll
      for (int j = 0; j < 4; ++j) o[j].f[i] *= scale;
    }

    // O += P @ V  (P tile is wave-private in LDS)
#pragma unroll
    for (int kt = 0; kt < 2; ++kt) {
      v16bf a = load_A_frag(Ps[wave], 0, kt * 32, LDK, lane);
#pragma unroll
      for (int j = 0; j < 4; ++j) {
        v16bf bb = load_Bt_frag(Vt, j * 16, kt * 32, LDK, lane);
        o[j].v = wmma_bf16(a, bb, o[j].v);
      }
    }
  }

  // write O (bf16) in [B, S, H*64] layout for the output GEMM
#pragma unroll
  for (int j = 0; j < 4; ++j)
#pragma unroll
    for (int i = 0; i < 8; ++i) {
      int qg = q0 + wave * 16 + mloc0 + i;
      int d  = j * 16 + nloc;
      size_t off = ((size_t)b * SEQ + qg) * INNER + h * D_KV + d;
      Ob[off] = f32_to_bf16_rn(o[j].f[i] / rl[i]);
    }
}

// ============================================================
// Kernel 3: output projection.  out = O[8192,384] @ Wo[384,1472] (f32 out)
// A tile (already bf16) staged via async DMA to LDS.
// ============================================================
__global__ void __launch_bounds__(128)
out_gemm_kernel(const unsigned short* __restrict__ Ob,
                const float* __restrict__ Wo, float* __restrict__ out) {
  __shared__ unsigned short As[BM * LDA];
  __shared__ unsigned short Bt[BN * LDA];

  const int tid  = threadIdx.x;
  const int lane = tid & 31;
  const int wave = tid >> 5;
  const int wm = (wave >> 1) * 32;
  const int wn = (wave & 1) * 32;
  const int m0 = blockIdx.y * BM;
  const int n0 = blockIdx.x * BN;

  AccF acc[2][2];
#pragma unroll
  for (int ti = 0; ti < 2; ++ti)
#pragma unroll
    for (int tj = 0; tj < 2; ++tj)
#pragma unroll
      for (int i = 0; i < 8; ++i) acc[ti][tj].f[i] = 0.0f;

  for (int k0 = 0; k0 < INNER; k0 += BK) {
    __syncthreads();
    if (k0 + BK < INNER)
      __builtin_prefetch(&Wo[(size_t)(k0 + BK + (tid >> 2)) * D_MODEL + n0], 0, 3);
    // stage A (already bf16) via async DMA: 64x32, 256 x 16B transfers
#pragma unroll
    for (int it = 0; it < 2; ++it) {
      int i = tid + it * 128;
      int r = i >> 2, c8 = (i & 3) << 3;
      async_cp_b128(&Ob[(size_t)(m0 + r) * INNER + k0 + c8], &As[r * LDA + c8]);
    }
    // stage B^T (Wo f32 -> bf16 transposed)
#pragma unroll
    for (int it = 0; it < 4; ++it) {
      int i = tid + it * 128;
      int k  = i >> 4;
      int n4 = (i & 15) << 2;
      F128 f = *(const F128*)&Wo[(size_t)(k0 + k) * D_MODEL + n0 + n4];
      Bt[(n4 + 0) * LDA + k] = f32_to_bf16_rn(f.x);
      Bt[(n4 + 1) * LDA + k] = f32_to_bf16_rn(f.y);
      Bt[(n4 + 2) * LDA + k] = f32_to_bf16_rn(f.z);
      Bt[(n4 + 3) * LDA + k] = f32_to_bf16_rn(f.w);
    }
    async_wait_all();
    __syncthreads();

    v16bf af[2];
    af[0] = load_A_frag(As, wm + 0,  0, LDA, lane);
    af[1] = load_A_frag(As, wm + 16, 0, LDA, lane);
#pragma unroll
    for (int tj = 0; tj < 2; ++tj) {
      v16bf bf = load_Bt_frag(Bt, wn + tj * 16, 0, LDA, lane);
#pragma unroll
      for (int ti = 0; ti < 2; ++ti)
        acc[ti][tj].v = wmma_bf16(af[ti], bf, acc[ti][tj].v);
    }
  }

  const int mloc0 = (lane >> 4) << 3;
  const int nloc  = lane & 15;
#pragma unroll
  for (int ti = 0; ti < 2; ++ti)
#pragma unroll
    for (int tj = 0; tj < 2; ++tj)
#pragma unroll
      for (int i = 0; i < 8; ++i) {
        int m = m0 + wm + ti * 16 + mloc0 + i;
        int n = n0 + wn + tj * 16 + nloc;
        out[(size_t)m * D_MODEL + n] = acc[ti][tj].f[i];
      }
}

// ============================================================
extern "C" void kernel_launch(void* const* d_in, const int* in_sizes, int n_in,
                              void* d_out, int out_size, void* d_ws, size_t ws_size,
                              hipStream_t stream) {
  const float* x   = (const float*)d_in[0];
  const float* Wq  = (const float*)d_in[1];
  const float* Wk  = (const float*)d_in[2];
  const float* Wv  = (const float*)d_in[3];
  const float* Wo  = (const float*)d_in[4];
  const float* rel = (const float*)d_in[5];
  float* out = (float*)d_out;

  const size_t perMat = (size_t)BATCH * N_HEADS * SEQ * D_KV;  // 3,145,728
  unsigned short* qb = (unsigned short*)d_ws;
  unsigned short* kb = qb + perMat;
  unsigned short* vb = kb + perMat;
  unsigned short* ob = vb + perMat;                            // [B,S,384] bf16

  dim3 g1(INNER / BN, (BATCH * SEQ) / BM);     // 6 x 128
  qkv_kernel<<<g1, 128, 0, stream>>>(x, Wq, Wk, Wv, qb, kb, vb);

  dim3 g2(BATCH * N_HEADS, SEQ / 64);          // 24 x 32
  attn_kernel<<<g2, 128, 0, stream>>>(qb, kb, vb, rel, ob);

  dim3 g3(D_MODEL / BN, (BATCH * SEQ) / BM);   // 23 x 128
  out_gemm_kernel<<<g3, 128, 0, stream>>>(ob, Wo, out);
}
